// PolitifactModel_76931454206414
// MI455X (gfx1250) — matrix operand order repaired and checked by reading, not compiled, not run on hardware
//
#include <hip/hip_runtime.h>
#include <hip/hip_bf16.h>
#include <stdint.h>

// ---------- types ----------
typedef __attribute__((ext_vector_type(8)))  float        v8f;
typedef __attribute__((ext_vector_type(16))) __bf16       v16bf;
typedef __attribute__((ext_vector_type(4)))  unsigned int u32x4;
typedef __attribute__((ext_vector_type(8)))  int          i32x8;
typedef __attribute__((ext_vector_type(4)))  int          i32x4;

union Frag {
  v16bf v;
  u32x4 q[2];
};

#define IN_F 768
#define H_F  128
#define QW   (IN_F / 2)       // 384 packed bf16x2 (u32) per row
#define KT   (IN_F / 32)      // 24 K-tiles
#define NEG_SLOPE 0.2f

// round-to-nearest-even f32 -> bf16, packed pair (lo = even K, hi = odd K)
__device__ inline unsigned int pack_bf16x2(float lo, float hi) {
  unsigned int a = __float_as_uint(lo);
  unsigned int b = __float_as_uint(hi);
  unsigned int alo = (a + 0x7FFFu + ((a >> 16) & 1u)) >> 16;
  unsigned int ahi = (b + 0x7FFFu + ((b >> 16) & 1u)) >> 16;
  return (ahi << 16) | (alo & 0xFFFFu);
}

// float atomic max: signed-int max for v>=0, unsigned-int min for v<0.
// Valid for cells initialized to -inf.
__device__ inline void atomicMaxF(float* addr, float v) {
  if (v >= 0.0f) {
    atomicMax((int*)addr, __float_as_int(v));
  } else {
    atomicMin((unsigned int*)addr, (unsigned int)__float_as_int(v));
  }
}

// ---------------------------------------------------------------------------
// Tensor Data Mover: 2D tile load (u32 elements) per CDNA5 ISA D# layout.
//   group0: count=1 | lds_addr | global_addr | type=2
//   group1: data_size=4B | tensor_dim0/1 | tile_dim0/1 | tensor_dim0_stride
//   groups2/3: zero (<=2D tile)
// dims/tiles/stride in u32 (data_size) units.
// ---------------------------------------------------------------------------
__device__ inline void tdm_load_2d(unsigned int lds_addr,
                                   unsigned long long gaddr,
                                   unsigned int dim0, unsigned int dim1,
                                   unsigned int tile0, unsigned int tile1,
                                   unsigned int stride0) {
  u32x4 g0;
  g0.x = 1u;                                              // count=1 (valid)
  g0.y = lds_addr;                                        // bits 63:32
  g0.z = (unsigned int)(gaddr & 0xFFFFFFFFull);           // addr[31:0]
  g0.w = (unsigned int)((gaddr >> 32) & 0x01FFFFFFull)    // addr[56:32]
         | (2u << 30);                                    // type=2 ("image")
  i32x8 g1;
  g1[0] = (int)(2u << 16);                                // data_size=4B
  g1[1] = (int)((dim0 & 0xFFFFu) << 16);                  // tensor_dim0 lo
  g1[2] = (int)((dim0 >> 16) | ((dim1 & 0xFFFFu) << 16)); // dim0 hi | dim1 lo
  g1[3] = (int)((dim1 >> 16) | (tile0 << 16));            // dim1 hi | tile_dim0
  g1[4] = (int)(tile1 & 0xFFFFu);                         // tile_dim1 | tile_dim2=0
  g1[5] = (int)stride0;                                   // tensor_dim0_stride[31:0]
  g1[6] = 0;                                              // stride hi | dim1_stride lo
  g1[7] = 0;
  i32x4 z4 = {0, 0, 0, 0};
#if defined(__clang_major__) && (__clang_major__ >= 23)
  i32x8 z8 = {0, 0, 0, 0, 0, 0, 0, 0};
  __builtin_amdgcn_tensor_load_to_lds(g0, g1, z4, z4, z8, 0);
#else
  __builtin_amdgcn_tensor_load_to_lds(g0, g1, z4, z4, 0);
#endif
}

// ---------------------------------------------------------------------------
// Pre-pass: x [N][768] f32 -> xbq [N][384] packed bf16x2 (row-major)
// ---------------------------------------------------------------------------
__global__ void k_cvt_x(const float* __restrict__ x, unsigned int* __restrict__ xbq,
                        long long total) {
  const long long i = (long long)blockIdx.x * blockDim.x + threadIdx.x;
  if (i >= total) return;
  float2 f = ((const float2*)x)[i];
  xbq[i] = pack_bf16x2(f.x, f.y);
}

// ---------------------------------------------------------------------------
// Pre-pass: W [768][128] f32 -> wbq [128][384]: wbq[n*384+q] = pack(W[2q][n], W[2q+1][n])
// (n-major, q contiguous => B LDS tile is a plain 2D TDM tile)
// ---------------------------------------------------------------------------
__global__ void k_cvt_w(const float* __restrict__ W, unsigned int* __restrict__ wbq) {
  const int idx = blockIdx.x * blockDim.x + threadIdx.x;
  if (idx >= H_F * QW) return;
  const int n = idx / QW, q = idx - n * QW;
  wbq[idx] = pack_bf16x2(W[(size_t)(2 * q) * H_F + n], W[(size_t)(2 * q + 1) * H_F + n]);
}

// ---------------------------------------------------------------------------
// Kernel 1: xp = x @ W  via v_wmma_f32_16x16x32_bf16, TDM-staged LDS tiles.
// Block: 256 threads = 8 waves; block tile 64 rows x 128 cols.
// Wave w: row-tile rt = w>>1 (16 rows), col-half ch = w&1 (4 x 16-col tiles).
// Double-buffered: wave 0 issues tensor_load_to_lds for tile t+1 while all
// waves run WMMA on tile t; sync = s_wait_tensorcnt(0) + workgroup barrier.
// Fragments: all 10 ds_load_b128 issued up front, then 4 independent WMMAs
// back-to-back (no D->A/B RAW => no mandated hazard nops between them).
// ---------------------------------------------------------------------------
__global__ __launch_bounds__(256) void k_gemm_xw(
    const unsigned int* __restrict__ xbq, const unsigned int* __restrict__ wbq,
    float* __restrict__ xp, int N) {
  __shared__ unsigned int at[2][64 * 16];
  __shared__ unsigned int bt[2][128 * 16];

  const int tid  = threadIdx.x;
  const int wave = tid >> 5;
  const int lane = tid & 31;
  const int rt   = wave >> 1;
  const int ch   = wave & 1;
  const int rowBase = blockIdx.x * 64;
  const unsigned int rowsLeft = (unsigned int)(N - rowBase);

  // lane-dependent LDS fragment offsets (u32 units), hoisted out of K loop
  const int aoff = (rt * 16 + (lane & 15)) * 16 + ((lane < 16) ? 0 : 8);
  const int boff = (ch * 64 + (lane & 15)) * 16 + ((lane < 16) ? 0 : 8);

  v8f acc[4] = {};

  if (wave == 0) {
    // tile 0: A then B
    tdm_load_2d((unsigned int)(uintptr_t)&at[0][0],
                (unsigned long long)(uintptr_t)(xbq + (size_t)rowBase * QW),
                QW, rowsLeft, 16, 64, QW);
    tdm_load_2d((unsigned int)(uintptr_t)&bt[0][0],
                (unsigned long long)(uintptr_t)wbq,
                QW, H_F, 16, 128, QW);
  }

  for (int t = 0; t < KT; ++t) {
    if (wave == 0) __builtin_amdgcn_s_wait_tensorcnt(0);
    __syncthreads();                       // tile t resident for all waves
    if (wave == 0 && (t + 1) < KT) {       // prefetch tile t+1 into other buffer
      const int q0 = (t + 1) * 16;
      const int buf = (t + 1) & 1;
      tdm_load_2d((unsigned int)(uintptr_t)&at[buf][0],
                  (unsigned long long)(uintptr_t)(xbq + (size_t)rowBase * QW + q0),
                  QW - q0, rowsLeft, 16, 64, QW);
      tdm_load_2d((unsigned int)(uintptr_t)&bt[buf][0],
                  (unsigned long long)(uintptr_t)(wbq + q0),
                  QW - q0, H_F, 16, 128, QW);
    }

    const unsigned int* A = &at[t & 1][0] + aoff;
    const unsigned int* B = &bt[t & 1][0] + boff;

    // load ALL fragments first: 10x ds_load_b128
    Frag a, b[4];
    a.q[0] = *(const u32x4*)(A);
    a.q[1] = *(const u32x4*)(A + 4);
    #pragma unroll
    for (int ct = 0; ct < 4; ++ct) {
      b[ct].q[0] = *(const u32x4*)(B + ct * 256);
      b[ct].q[1] = *(const u32x4*)(B + ct * 256 + 4);
    }
    // then 4 independent WMMAs back-to-back
    #pragma unroll
    for (int ct = 0; ct < 4; ++ct) {
      acc[ct] = __builtin_amdgcn_wmma_f32_16x16x32_bf16(
          false, a.v, false, b[ct].v, (short)0, acc[ct], false, false);
    }
    __syncthreads();                       // all reads of tile t done
  }

  // store C: VGPR v -> row rt*16 + v (+8 for upper half-wave), col = lane&15
  #pragma unroll
  for (int ct = 0; ct < 4; ++ct) {
    const int col = ch * 64 + ct * 16 + (lane & 15);
    const int rb  = rowBase + rt * 16 + ((lane < 16) ? 0 : 8);
    #pragma unroll
    for (int v = 0; v < 8; ++v) {
      const int row = rb + v;
      if (row < N) xp[(size_t)row * H_F + col] = acc[ct][v];
    }
  }
}

// ---------------------------------------------------------------------------
// Kernel 2: al_src[n] = xp[n,:]·a_src ; al_dst[n] = xp[n,:]·a_dst  (1 wave/node)
// ---------------------------------------------------------------------------
__global__ __launch_bounds__(256) void k_al(
    const float* __restrict__ xp, const float* __restrict__ a_src,
    const float* __restrict__ a_dst, float* __restrict__ al_src,
    float* __restrict__ al_dst, int N) {
  const int node = (blockIdx.x * blockDim.x + threadIdx.x) >> 5;
  const int lane = threadIdx.x & 31;
  if (node >= N) return;
  float4 v  = ((const float4*)(xp + (size_t)node * H_F))[lane];
  float4 as = ((const float4*)a_src)[lane];
  float4 ad = ((const float4*)a_dst)[lane];
  float ss = v.x * as.x + v.y * as.y + v.z * as.z + v.w * as.w;
  float sd = v.x * ad.x + v.y * ad.y + v.z * ad.z + v.w * ad.w;
  #pragma unroll
  for (int m = 16; m >= 1; m >>= 1) {
    ss += __shfl_xor(ss, m, 32);
    sd += __shfl_xor(sd, m, 32);
  }
  if (lane == 0) { al_src[node] = ss; al_dst[node] = sd; }
}

// ---------------------------------------------------------------------------
// Init: outconv=0, emax=-inf, denom=0, pool=0
// ---------------------------------------------------------------------------
__global__ void k_init(float* outconv, float* emax, float* denom, float* pool,
                       int N, int G) {
  const size_t total = (size_t)N * H_F;
  for (size_t i = (size_t)blockIdx.x * blockDim.x + threadIdx.x; i < total;
       i += (size_t)gridDim.x * blockDim.x) {
    outconv[i] = 0.0f;
    if (i < (size_t)N) { emax[i] = -__builtin_inff(); denom[i] = 0.0f; }
    if (i < (size_t)G * H_F) pool[i] = 0.0f;
  }
}

// ---------------------------------------------------------------------------
// Edge pass 1: e = leaky_relu(al_src[s] + al_dst[d]); segment max over dst
// ---------------------------------------------------------------------------
__global__ void k_edge1(const int* __restrict__ ei, const float* __restrict__ al_src,
                        const float* __restrict__ al_dst, float* __restrict__ e_buf,
                        float* __restrict__ emax, int E, int N) {
  const int i = blockIdx.x * blockDim.x + threadIdx.x;
  if (i >= E + N) return;
  int s, d;
  if (i < E) { s = ei[i]; d = ei[E + i]; } else { s = d = i - E; }
  float e = al_src[s] + al_dst[d];
  e = (e > 0.0f) ? e : NEG_SLOPE * e;
  e_buf[i] = e;
  atomicMaxF(&emax[d], e);
}

// ---------------------------------------------------------------------------
// Edge pass 2: ee = exp(e - emax[dst]); segment sum -> denom
// ---------------------------------------------------------------------------
__global__ void k_edge2(const int* __restrict__ ei, float* __restrict__ e_buf,
                        const float* __restrict__ emax, float* __restrict__ denom,
                        int E, int N) {
  const int i = blockIdx.x * blockDim.x + threadIdx.x;
  if (i >= E + N) return;
  int d = (i < E) ? ei[E + i] : (i - E);
  float ee = __expf(e_buf[i] - emax[d]);
  e_buf[i] = ee;
  atomicAdd(&denom[d], ee);
}

// ---------------------------------------------------------------------------
// Edge pass 3: outconv[d,:] += (ee/denom[d]) * xp[s,:]   (1 wave per edge)
// ---------------------------------------------------------------------------
__global__ __launch_bounds__(256) void k_edge3(
    const int* __restrict__ ei, const float* __restrict__ e_buf,
    const float* __restrict__ denom, const float* __restrict__ xp,
    float* __restrict__ outconv, int E, int N) {
  const int idx  = blockIdx.x * 8 + (threadIdx.x >> 5);
  const int lane = threadIdx.x & 31;
  if (idx >= E + N) return;
  int s, d;
  if (idx < E) { s = ei[idx]; d = ei[E + idx]; } else { s = d = idx - E; }
  const float alpha = e_buf[idx] / (denom[d] + 1e-16f);
  float4 v = ((const float4*)(xp + (size_t)s * H_F))[lane];
  float* o = outconv + (size_t)d * H_F + lane * 4;
  atomicAdd(o + 0, alpha * v.x);
  atomicAdd(o + 1, alpha * v.y);
  atomicAdd(o + 2, alpha * v.z);
  atomicAdd(o + 3, alpha * v.w);
}

// ---------------------------------------------------------------------------
// ReLU + per-graph max pool (relu >= 0 and pool init 0 => int atomicMax ok)
// ---------------------------------------------------------------------------
__global__ void k_pool(const float* __restrict__ outconv, const float* __restrict__ b_conv,
                       const int* __restrict__ batch, float* __restrict__ pool, int N) {
  const size_t i = (size_t)blockIdx.x * blockDim.x + threadIdx.x;
  if (i >= (size_t)N * H_F) return;
  const int c = (int)(i & (H_F - 1));
  const size_t n = i >> 7;
  float r = fmaxf(outconv[i] + b_conv[c], 0.0f);
  atomicMax((int*)&pool[(size_t)batch[n] * H_F + c], __float_as_int(r));
}

// ---------------------------------------------------------------------------
// roots[g] = first node of graph g (batch sorted)
// ---------------------------------------------------------------------------
__global__ void k_roots(const int* __restrict__ batch, int* __restrict__ roots, int N) {
  const int i = blockIdx.x * blockDim.x + threadIdx.x;
  if (i >= N) return;
  const int b = batch[i];
  if (i == 0 || batch[i - 1] != b) roots[b] = i;
}

// ---------------------------------------------------------------------------
// h_lin = relu(x[roots] @ W0 + b0)   (one block of 128 threads per graph)
// ---------------------------------------------------------------------------
__global__ __launch_bounds__(128) void k_lin(
    const float* __restrict__ x, const int* __restrict__ roots,
    const float* __restrict__ W0, const float* __restrict__ b0,
    float* __restrict__ hlin) {
  const int g = blockIdx.x, t = threadIdx.x;
  const float* xr = x + (size_t)roots[g] * IN_F;
  float acc = b0[t];
  for (int k = 0; k < IN_F; ++k) acc = fmaf(xr[k], W0[(size_t)k * H_F + t], acc);
  hlin[(size_t)g * H_F + t] = fmaxf(acc, 0.0f);
}

// ---------------------------------------------------------------------------
// Tail MLP: h = [pool | hlin]; out = relu(h@W1+b1) @ W2 + b2
// ---------------------------------------------------------------------------
__global__ __launch_bounds__(128) void k_mlp(
    const float* __restrict__ pool, const float* __restrict__ hlin,
    const float* __restrict__ W1, const float* __restrict__ b1,
    const float* __restrict__ W2, const float* __restrict__ b2,
    float* __restrict__ out) {
  __shared__ float h[2 * H_F];
  __shared__ float h1[H_F];
  const int g = blockIdx.x, t = threadIdx.x;
  h[t]        = pool[(size_t)g * H_F + t];
  h[H_F + t]  = hlin[(size_t)g * H_F + t];
  __syncthreads();
  float acc = b1[t];
  for (int j = 0; j < 2 * H_F; ++j) acc = fmaf(h[j], W1[(size_t)j * H_F + t], acc);
  h1[t] = fmaxf(acc, 0.0f);
  __syncthreads();
  if (t < 2) {
    float o = b2[t];
    for (int j = 0; j < H_F; ++j) o = fmaf(h1[j], W2[(size_t)j * 2 + t], o);
    out[(size_t)g * 2 + t] = o;
  }
}

// ---------------------------------------------------------------------------
extern "C" void kernel_launch(void* const* d_in, const int* in_sizes, int n_in,
                              void* d_out, int out_size, void* d_ws, size_t ws_size,
                              hipStream_t stream) {
  // dict order: x, edge_index, batch, W, a_src, a_dst, b_conv, W0, b0, W1, b1, W2, b2
  const float* x      = (const float*)d_in[0];
  const int*   ei     = (const int*)d_in[1];
  const int*   batch  = (const int*)d_in[2];
  const float* W      = (const float*)d_in[3];
  const float* a_src  = (const float*)d_in[4];
  const float* a_dst  = (const float*)d_in[5];
  const float* b_conv = (const float*)d_in[6];
  const float* W0     = (const float*)d_in[7];
  const float* b0     = (const float*)d_in[8];
  const float* W1     = (const float*)d_in[9];
  const float* b1     = (const float*)d_in[10];
  const float* W2     = (const float*)d_in[11];
  const float* b2     = (const float*)d_in[12];
  float* out = (float*)d_out;

  const int N = in_sizes[0] / IN_F;   // 50000
  const int E = in_sizes[1] / 2;      // 800000
  const int G = out_size / 2;         // 500
  const int EN = E + N;

  // workspace carve-up
  float* ws = (float*)d_ws;
  float* xp      = ws; ws += (size_t)N * H_F;
  float* outconv = ws; ws += (size_t)N * H_F;
  float* al_src  = ws; ws += N;
  float* al_dst  = ws; ws += N;
  float* emax    = ws; ws += N;
  float* denom   = ws; ws += N;
  float* e_buf   = ws; ws += EN;
  float* pool    = ws; ws += (size_t)G * H_F;
  float* hlin    = ws; ws += (size_t)G * H_F;
  int*   roots   = (int*)ws; ws += G;
  unsigned int* xbq = (unsigned int*)ws; ws += (size_t)N * QW;
  unsigned int* wbq = (unsigned int*)ws;

  const long long xq_total = (long long)N * QW;
  k_cvt_x<<<(unsigned)((xq_total + 255) / 256), 256, 0, stream>>>(x, xbq, xq_total);
  k_cvt_w<<<(H_F * QW + 255) / 256, 256, 0, stream>>>(W, wbq);
  k_init<<<2048, 256, 0, stream>>>(outconv, emax, denom, pool, N, G);
  k_gemm_xw<<<(N + 63) / 64, 256, 0, stream>>>(xbq, wbq, xp, N);
  k_al<<<(N + 7) / 8, 256, 0, stream>>>(xp, a_src, a_dst, al_src, al_dst, N);
  k_roots<<<(N + 255) / 256, 256, 0, stream>>>(batch, roots, N);
  k_edge1<<<(EN + 255) / 256, 256, 0, stream>>>(ei, al_src, al_dst, e_buf, emax, E, N);
  k_edge2<<<(EN + 255) / 256, 256, 0, stream>>>(ei, e_buf, emax, denom, E, N);
  k_edge3<<<(EN + 7) / 8, 256, 0, stream>>>(ei, e_buf, denom, xp, outconv, E, N);
  {
    const size_t total = (size_t)N * H_F;
    k_pool<<<(unsigned)((total + 255) / 256), 256, 0, stream>>>(outconv, b_conv, batch, pool, N);
  }
  k_lin<<<G, 128, 0, stream>>>(x, roots, W0, b0, hlin);
  k_mlp<<<G, 128, 0, stream>>>(pool, hlin, W1, b1, W2, b2, out);
}
